// NeuronLevelModelsConv_15685220565446
// MI455X (gfx1250) — matrix operand order repaired and checked by reading, not compiled, not run on hardware
//
#include <hip/hip_runtime.h>
#include <stdint.h>

// Sizes from the reference
#define B_   512
#define D_   2048
#define M_   128
#define H_   128
#define BT   128        // rows of B handled per workgroup
#define LDSW 136        // padded LDS row stride (bf16 elems): rows step 4 banks -> conflict-free

typedef __attribute__((ext_vector_type(16))) __bf16 v16bf;
typedef __attribute__((ext_vector_type(8)))  float  v8f;
typedef __attribute__((ext_vector_type(4)))  float  v4f;   // clang ext-vector: valid for nontemporal builtins

union ABRegs { v16bf v; __bf16 b[16]; uint4 q[2]; };
union Pack4 { __bf16 b[4]; uint2 u; };

// Hardware tanh on gfx1250 (TRANS32 op); fall back to libm if the builtin is absent.
__device__ __forceinline__ float hw_tanhf(float x) {
#if __has_builtin(__builtin_amdgcn_tanh_f32)
  return __builtin_amdgcn_tanh_f32(x);
#elif __has_builtin(__builtin_amdgcn_tanhf)
  return __builtin_amdgcn_tanhf(x);
#else
  return tanhf(x);
#endif
}

// GELU, tanh form: 0.5*x*(1+tanh(sqrt(2/pi)*(x + 0.044715 x^3)))
// |error vs exact-erf gelu| ~1e-3 abs, far below the bf16 GEMM quantization noise.
__device__ __forceinline__ float gelu_fast(float x) {
  float u  = x * x;
  float in = x * fmaf(0.0356774081f, u, 0.7978845608f);  // sqrt(2/pi)*(x + 0.044715x^3)
  float th = hw_tanhf(in);
  return x * fmaf(0.5f, th, 0.5f);
}

__global__ __launch_bounds__(256)
void neuron_mlp_kernel(const float* __restrict__ X,  const float* __restrict__ W1,
                       const float* __restrict__ b1, const float* __restrict__ W2,
                       const float* __restrict__ b2, float* __restrict__ out)
{
  __shared__ unsigned short sW[H_ * LDSW];   // W1[d] as bf16, padded rows (34 KB)

  const int d     = blockIdx.x;
  const int btile = blockIdx.y;
  const int tid   = threadIdx.x;
  const int lane  = tid & 31;
  const int wave  = tid >> 5;

  // ---------- stage W1[d] (128x128 f32) -> LDS bf16 (v_cvt_pk_bf16_f32) ----------
  const float* W1d = W1 + (size_t)d * (H_ * M_);
  #pragma unroll
  for (int it = 0; it < 16; ++it) {
    int idx = it * 256 + tid;          // float4 index, 32 per row
    int row = idx >> 5;
    int col = (idx & 31) << 2;
    v4f v = *reinterpret_cast<const v4f*>(W1d + row * M_ + col);
    Pack4 p;
    p.b[0] = (__bf16)v[0]; p.b[1] = (__bf16)v[1];
    p.b[2] = (__bf16)v[2]; p.b[3] = (__bf16)v[3];
    *reinterpret_cast<uint2*>(&sW[row * LDSW + col]) = p.u;
  }
  __syncthreads();

  // ---------- per-wave GEMM: 16 rows x 128 cols, K=128 ----------
  const int rowbase = btile * BT + wave * 16;
  const int halfsel = lane >> 4;          // 0 | 1
  const int l15     = lane & 15;

  // A operand (ISA 16-bit A layout): lane holds row (l15),
  // K = kb..kb+7 and kb+16..kb+23 where kb = halfsel*8, within each 32-chunk.
  const float* xrow = X + ((size_t)(rowbase + l15) * D_ + d) * M_;
  const int akb = halfsel << 3;

  // B operand (ISA 16-bit B layout): lane = column (l15),
// K = kb..kb+7 and kb+16..kb+23 where kb = halfsel*8 (same order as A).
  const int bBase = l15 * LDSW + (halfsel << 3);

  v8f Cacc[8] = {};

  #pragma unroll
  for (int kc = 0; kc < 4; ++kc) {
    const float* xp = xrow + kc * 32 + akb;
    v4f f0 = __builtin_nontemporal_load(reinterpret_cast<const v4f*>(xp));
    v4f f1 = __builtin_nontemporal_load(reinterpret_cast<const v4f*>(xp + 4));
    v4f f2 = __builtin_nontemporal_load(reinterpret_cast<const v4f*>(xp + 16));
    v4f f3 = __builtin_nontemporal_load(reinterpret_cast<const v4f*>(xp + 20));
    ABRegs A;
    A.b[0]  = (__bf16)f0[0]; A.b[1]  = (__bf16)f0[1];
    A.b[2]  = (__bf16)f0[2]; A.b[3]  = (__bf16)f0[3];
    A.b[4]  = (__bf16)f1[0]; A.b[5]  = (__bf16)f1[1];
    A.b[6]  = (__bf16)f1[2]; A.b[7]  = (__bf16)f1[3];
    A.b[8]  = (__bf16)f2[0]; A.b[9]  = (__bf16)f2[1];
    A.b[10] = (__bf16)f2[2]; A.b[11] = (__bf16)f2[3];
    A.b[12] = (__bf16)f3[0]; A.b[13] = (__bf16)f3[1];
    A.b[14] = (__bf16)f3[2]; A.b[15] = (__bf16)f3[3];

    #pragma unroll
    for (int nt = 0; nt < 8; ++nt) {
      ABRegs Bb;
      const unsigned short* bp = &sW[nt * 16 * LDSW + bBase + kc * 32];
      Bb.q[0] = *reinterpret_cast<const uint4*>(bp);
      Bb.q[1] = *reinterpret_cast<const uint4*>(bp + 16);
      Cacc[nt] = __builtin_amdgcn_wmma_f32_16x16x32_bf16(
          false, A.v, false, Bb.v, (short)0, Cacc[nt], false, false);
    }
  }

  // ---------- epilogue: +b1, GELU (hw tanh), *W2, reduce over H ----------
  const float b2d = b2[d];
  float acc[8];
  #pragma unroll
  for (int r = 0; r < 8; ++r) acc[r] = 0.f;

  #pragma unroll
  for (int nt = 0; nt < 8; ++nt) {
    const int hcol = nt * 16 + l15;
    const float bb = b1[(size_t)d * H_ + hcol];
    const float ww = W2[(size_t)d * H_ + hcol];
    #pragma unroll
    for (int r = 0; r < 8; ++r) {
      float hv = Cacc[nt][r] + bb;                       // C layout: M=r(+8), N=l15
      acc[r] = fmaf(gelu_fast(hv), ww, acc[r]);
    }
  }

  #pragma unroll
  for (int r = 0; r < 8; ++r) {
    float s = acc[r];
    s += __shfl_xor(s, 1, 32);
    s += __shfl_xor(s, 2, 32);
    s += __shfl_xor(s, 4, 32);
    s += __shfl_xor(s, 8, 32);   // sum within each 16-lane half
    acc[r] = s;
  }

  if (l15 == 0) {
    const int mofs = halfsel * 8;
    #pragma unroll
    for (int r = 0; r < 8; ++r) {
      __builtin_nontemporal_store(acc[r] + b2d,
          out + (size_t)(rowbase + mofs + r) * D_ + d);
    }
  }
}

extern "C" void kernel_launch(void* const* d_in, const int* in_sizes, int n_in,
                              void* d_out, int out_size, void* d_ws, size_t ws_size,
                              hipStream_t stream) {
  const float* X  = (const float*)d_in[0];
  const float* W1 = (const float*)d_in[1];
  const float* b1 = (const float*)d_in[2];
  const float* W2 = (const float*)d_in[3];
  const float* b2 = (const float*)d_in[4];
  float* out = (float*)d_out;

  dim3 grid(D_, B_ / BT);   // 2048 x 4 workgroups
  dim3 block(256);          // 8 waves
  neuron_mlp_kernel<<<grid, block, 0, stream>>>(X, W1, b1, W2, b2, out);
}
